// MambaDecoderLayer_28638841929752
// MI455X (gfx1250) — matrix operand order repaired and checked by reading, not compiled, not run on hardware
//
#include <hip/hip_runtime.h>

// ---------------- problem constants ----------------
#define DM    2048      // d_model
#define DI    4096      // d_inner
#define DXB   1024      // d_xb
#define DST   16        // d_state
#define DCONV 4
#define DTR   128       // dt_rank
#define INTER 8192
#define PROJ  (2*DI + 2*DXB + DTR)   // 10368
#define BATCH 2
#define SEQ   1024
#define TOK   (BATCH*SEQ)            // 2048

// zxbcdt column offsets
#define OZ  0
#define OX  DI            // 4096
#define OB  (DI+DXB)      // 5120
#define OC  (DI+2*DXB)    // 6144
#define ODT (2*DI+2*DXB)  // 10240

typedef _Float16 half8  __attribute__((ext_vector_type(8)));
typedef _Float16 v16h   __attribute__((ext_vector_type(16)));
typedef float    v8f    __attribute__((ext_vector_type(8)));

// ---------------- WMMA fragment loads (wave32, gfx1250; this code's contiguous k order, same for A and B) ----------------
// A (16x32 f16): lane r=lane&15 -> row M=r ; lg=lane>>4 -> halves = K{lg*16..+15} contiguous
__device__ __forceinline__ v16h load_a_frag(const _Float16* p) {
  half8 lo = *(const half8*)(p);
  half8 hi = *(const half8*)(p + 8);
  return __builtin_shufflevector(lo, hi, 0,1,2,3,4,5,6,7,8,9,10,11,12,13,14,15);
}
// B (32x16 f16, from row-major W[N][K] == B^T): lane r -> col N=r ; halves = K{lg*16..+15} contiguous
__device__ __forceinline__ v16h load_b_frag(const _Float16* p) {
  half8 lo = *(const half8*)(p);
  half8 hi = *(const half8*)(p + 8);
  return __builtin_shufflevector(lo, hi, 0,1,2,3,4,5,6,7,8,9,10,11,12,13,14,15);
}

// ---------------- tiled NT GEMM: C[M,N] = A[M,K] * W[N,K]^T (+resid) ----------------
// block = 256 threads = 8 waves (4 in M x 2 in N); wave tile 32x64; block tile 128x128
// K-loop is a 2-stage register pipeline with two named fragment sets alternating
// roles (no register copies): loads of step i+1 issue before the WMMAs of step i.
// Requires K % 64 == 0 (all K here: 2048, 128, 4096, 8192).
__global__ __launch_bounds__(256) void gemm_wmma_f16(
    const _Float16* __restrict__ A, const _Float16* __restrict__ W,
    float* __restrict__ C, const float* __restrict__ resid,
    int M, int N, int K)
{
  const int lane = threadIdx.x & 31;
  const int wv   = threadIdx.x >> 5;
  const int wm   = wv & 3;            // 4 waves across M
  const int wn   = wv >> 2;           // 2 waves across N
  const int m0   = blockIdx.y * 128 + wm * 32;
  const int n0   = blockIdx.x * 128 + wn * 64;
  const int lr   = lane & 15;
  const int lg   = lane >> 4;

  const _Float16* ap0 = A + (size_t)(m0 + lr) * K + lg * 16;
  const _Float16* ap1 = ap0 + (size_t)16 * K;
  const _Float16* bp0 = W + (size_t)(n0 + lr) * K + lg * 16;
  const _Float16* bp1 = bp0 + (size_t)16 * K;
  const _Float16* bp2 = bp0 + (size_t)32 * K;
  const _Float16* bp3 = bp0 + (size_t)48 * K;

  v8f acc[2][4];
#pragma unroll
  for (int i = 0; i < 2; ++i)
#pragma unroll
    for (int j = 0; j < 4; ++j) acc[i][j] = (v8f){};

  v16h aX[2], bX[4], aY[2], bY[4];

  auto loadX = [&](int kb) {
    aX[0] = load_a_frag(ap0 + kb); aX[1] = load_a_frag(ap1 + kb);
    bX[0] = load_b_frag(bp0 + kb); bX[1] = load_b_frag(bp1 + kb);
    bX[2] = load_b_frag(bp2 + kb); bX[3] = load_b_frag(bp3 + kb);
  };
  auto loadY = [&](int kb) {
    aY[0] = load_a_frag(ap0 + kb); aY[1] = load_a_frag(ap1 + kb);
    bY[0] = load_b_frag(bp0 + kb); bY[1] = load_b_frag(bp1 + kb);
    bY[2] = load_b_frag(bp2 + kb); bY[3] = load_b_frag(bp3 + kb);
  };
  auto wmX = [&]() {
#pragma unroll
    for (int i = 0; i < 2; ++i)
#pragma unroll
      for (int j = 0; j < 4; ++j)
        acc[i][j] = __builtin_amdgcn_wmma_f32_16x16x32_f16(
            false, aX[i], false, bX[j], (short)0, acc[i][j], false, false);
  };
  auto wmY = [&]() {
#pragma unroll
    for (int i = 0; i < 2; ++i)
#pragma unroll
      for (int j = 0; j < 4; ++j)
        acc[i][j] = __builtin_amdgcn_wmma_f32_16x16x32_f16(
            false, aY[i], false, bY[j], (short)0, acc[i][j], false, false);
  };

  // 2-stage pipeline over pairs of 32-wide k-steps
  loadX(0);
  int kb = 0;
  for (; kb + 64 < K; kb += 64) {
    loadY(kb + 32);
    wmX();
    loadX(kb + 64);
    wmY();
  }
  loadY(K - 32);
  wmX();
  wmY();

  // C/D f32 layout: VGPR r -> row base + lg*8 + r ; col base + lr
#pragma unroll
  for (int i = 0; i < 2; ++i) {
    const int mb = m0 + i * 16 + lg * 8;
#pragma unroll
    for (int j = 0; j < 4; ++j) {
      const int nc = n0 + j * 16 + lr;
#pragma unroll
      for (int r = 0; r < 8; ++r) {
        size_t idx = (size_t)(mb + r) * N + nc;
        C[idx] = (resid ? resid[idx] : 0.f) + acc[i][j][r];
      }
    }
  }
}

// ---------------- RMSNorm (fp32 in) -> f16 out ----------------
__global__ __launch_bounds__(256) void rmsnorm_cast_kernel(
    const float* __restrict__ x, const float* __restrict__ w,
    _Float16* __restrict__ y, int width)
{
  __shared__ float red[8];
  const int t = blockIdx.x;
  const float* row = x + (size_t)t * width;
  float ss = 0.f;
  for (int c = threadIdx.x; c < width; c += 256) { float v = row[c]; ss += v * v; }
#pragma unroll
  for (int off = 16; off; off >>= 1) ss += __shfl_xor(ss, off, 32);
  if ((threadIdx.x & 31) == 0) red[threadIdx.x >> 5] = ss;
  __syncthreads();
  if (threadIdx.x < 8) {
    float v = red[threadIdx.x];
#pragma unroll
    for (int off = 4; off; off >>= 1) v += __shfl_xor(v, off, 32);
    if (threadIdx.x == 0) red[0] = v;
  }
  __syncthreads();
  const float scale = rsqrtf(red[0] / (float)width + 1e-5f);
  for (int c = threadIdx.x; c < width; c += 256)
    y[(size_t)t * width + c] = (_Float16)(row[c] * scale * w[c]);
}

// ---------------- generic f32 -> f16 cast ----------------
__global__ void cast_f16_kernel(const float* __restrict__ s, _Float16* __restrict__ d, size_t n)
{
  size_t i = (size_t)blockIdx.x * 256 + threadIdx.x;
  if (i < n) d[i] = (_Float16)s[i];
}

// strided column-slice cast (dt columns of zxbcdt)
__global__ void cast_slice_kernel(const float* __restrict__ s, int ld, int col0,
                                  int width, _Float16* __restrict__ d)
{
  const int t = blockIdx.x;
  for (int c = threadIdx.x; c < width; c += blockDim.x)
    d[(size_t)t * width + c] = (_Float16)s[(size_t)t * ld + col0 + c];
}

// ---------------- causal depthwise conv1d (+repeat_kv gather) + silu ----------------
__global__ __launch_bounds__(256) void conv_silu_kernel(
    const float* __restrict__ zxbcdt, const float* __restrict__ cw,
    const float* __restrict__ cb, float* __restrict__ xc)
{
  const int t = blockIdx.x;
  const int d = blockIdx.y * 256 + threadIdx.x;     // channel in [0, DI)
  const int l = t & (SEQ - 1);
  const int b = t >> 10;
  const int xcol = ((d >> 6) << 4) | (d & 15);      // repeat_kv source column in x slice
  float acc = cb[d];
#pragma unroll
  for (int j = 0; j < DCONV; ++j) {
    int ll = l - (DCONV - 1) + j;
    if (ll >= 0)
      acc += cw[d * DCONV + j] * zxbcdt[(size_t)(b * SEQ + ll) * PROJ + OX + xcol];
  }
  xc[(size_t)t * DI + d] = acc / (1.f + __expf(-acc));   // silu
}

// ---------------- selective scan (sequential over L, states in VGPRs) ----------------
// grid.x = BATCH*16 ; one thread per channel d = chunk*256 + tid
__global__ __launch_bounds__(256) void scan_kernel(
    const float* __restrict__ zxbcdt, const float* __restrict__ dtraw,
    const float* __restrict__ xc, const float* __restrict__ A_log,
    const float* __restrict__ dt_b, const float* __restrict__ Dp,
    _Float16* __restrict__ y16)
{
  const int b     = blockIdx.x >> 4;
  const int chunk = blockIdx.x & 15;
  const int tid   = threadIdx.x;
  const int d     = chunk * 256 + tid;

  __shared__ float Cs[256];   // 16 heads x 16 states
  __shared__ float Bs[64];    // 4 source heads x 16 states

  float a[DST], h[DST];
#pragma unroll
  for (int n = 0; n < DST; ++n) { a[n] = -__expf(A_log[(size_t)d * DST + n]); h[n] = 0.f; }
  const float bias = dt_b[d];
  const float Dv   = Dp[d];
  const int hloc   = tid >> 4;          // local C head
  const int nlo    = tid & 15;
  const int shloc  = tid >> 6;          // local B source head

  for (int l = 0; l < SEQ; ++l) {
    const size_t t = (size_t)b * SEQ + l;
    const float* row = zxbcdt + t * PROJ;
    Cs[tid] = row[OC + (chunk * 16 + hloc) * DST + nlo];
    if (tid < 64) Bs[tid] = row[OB + (chunk * 4 + (tid >> 4)) * DST + nlo];
    __syncthreads();

    float dtv = dtraw[t * DI + d] + bias;
    dtv = (dtv > 20.f) ? dtv : log1pf(__expf(dtv));       // softplus
    const float xv  = xc[t * DI + d];
    const float dtx = dtv * xv;
    float y = 0.f;
#pragma unroll
    for (int n = 0; n < DST; ++n) {
      const float da = __expf(dtv * a[n]);
      h[n] = da * h[n] + dtx * Bs[shloc * DST + n];
      y += h[n] * Cs[hloc * DST + n];
    }
    const float z   = row[OZ + d];
    const float sil = z / (1.f + __expf(-z));
    y16[t * DI + d] = (_Float16)((y + xv * Dv) * sil);
    __syncthreads();
  }
}

// ---------------- silu(gate) * up -> f16 (gate/up stacked in one [TOK, 2*INTER] buffer) ----------------
__global__ void silu_mul_cast_kernel(const float* __restrict__ gu, _Float16* __restrict__ o)
{
  size_t i = (size_t)blockIdx.x * 256 + threadIdx.x;   // over TOK*INTER
  size_t t = i >> 13;                                  // / INTER
  size_t c = i & (INTER - 1);
  float gv = gu[t * (2 * INTER) + c];
  float uv = gu[t * (2 * INTER) + INTER + c];
  o[i] = (_Float16)((gv / (1.f + __expf(-gv))) * uv);
}

// ---------------- host orchestration ----------------
extern "C" void kernel_launch(void* const* d_in, const int* in_sizes, int n_in,
                              void* d_out, int out_size, void* d_ws, size_t ws_size,
                              hipStream_t stream) {
  (void)in_sizes; (void)n_in; (void)out_size; (void)ws_size;
  const float* hidden    = (const float*)d_in[0];
  const float* in_proj_w = (const float*)d_in[1];
  const float* conv_w    = (const float*)d_in[2];
  const float* conv_b    = (const float*)d_in[3];
  const float* dt_proj_w = (const float*)d_in[4];
  const float* dt_proj_b = (const float*)d_in[5];
  const float* A_log     = (const float*)d_in[6];
  const float* D_param   = (const float*)d_in[7];
  const float* out_proj_w= (const float*)d_in[8];
  const float* ln1       = (const float*)d_in[9];
  const float* ln2       = (const float*)d_in[10];
  const float* gate_w    = (const float*)d_in[11];
  const float* up_w      = (const float*)d_in[12];
  const float* down_w    = (const float*)d_in[13];
  float* out = (float*)d_out;

  char* ws = (char*)d_ws;
  size_t off = 0;
  auto alloc = [&](size_t bytes) -> char* {
    char* p = ws + off;
    off = (off + bytes + 255) & ~(size_t)255;
    return p;
  };

  _Float16* w_in16   = (_Float16*)alloc((size_t)PROJ * DM * 2);
  _Float16* w_dt16   = (_Float16*)alloc((size_t)DI * DTR * 2);
  _Float16* w_out16  = (_Float16*)alloc((size_t)DM * DI * 2);
  _Float16* w_gu16   = (_Float16*)alloc((size_t)2 * INTER * DM * 2);  // gate rows then up rows
  _Float16* w_down16 = (_Float16*)alloc((size_t)DM * INTER * 2);
  _Float16* hnorm16  = (_Float16*)alloc((size_t)TOK * DM * 2);
  float*    zxbcdt   = (float*)   alloc((size_t)TOK * PROJ * 4);
  _Float16* dt16     = (_Float16*)alloc((size_t)TOK * DTR * 2);
  float*    dtraw    = (float*)   alloc((size_t)TOK * DI * 4);
  float*    xconv    = (float*)   alloc((size_t)TOK * DI * 4);
  _Float16* y16      = (_Float16*)alloc((size_t)TOK * DI * 2);
  float*    residb   = (float*)   alloc((size_t)TOK * DM * 4);
  _Float16* h2_16    = (_Float16*)alloc((size_t)TOK * DM * 2);
  float*    gu       = (float*)   alloc((size_t)TOK * 2 * INTER * 4);
  _Float16* act16    = (_Float16*)alloc((size_t)TOK * INTER * 2);

  auto cast = [&](const float* s, _Float16* d, size_t n) {
    cast_f16_kernel<<<dim3((unsigned)((n + 255) / 256)), 256, 0, stream>>>(s, d, n);
  };
  auto gemm = [&](const _Float16* A, const _Float16* W, float* C,
                  const float* resid, int M, int N, int K) {
    dim3 grid((unsigned)(N / 128), (unsigned)(M / 128));
    gemm_wmma_f16<<<grid, 256, 0, stream>>>(A, W, C, resid, M, N, K);
  };

  // 1) cast weights to f16 (done fresh each call; deterministic)
  cast(in_proj_w,  w_in16,   (size_t)PROJ * DM);
  cast(dt_proj_w,  w_dt16,   (size_t)DI * DTR);
  cast(out_proj_w, w_out16,  (size_t)DM * DI);
  cast(gate_w,     w_gu16,                        (size_t)INTER * DM);
  cast(up_w,       w_gu16 + (size_t)INTER * DM,   (size_t)INTER * DM);
  cast(down_w,     w_down16, (size_t)DM * INTER);

  // 2) rmsnorm(hidden, ln1) -> f16
  rmsnorm_cast_kernel<<<TOK, 256, 0, stream>>>(hidden, ln1, hnorm16, DM);

  // 3) in_proj: [TOK,DM] x [PROJ,DM]^T -> zxbcdt [TOK,PROJ]
  gemm(hnorm16, w_in16, zxbcdt, nullptr, TOK, PROJ, DM);

  // 4) dt slice -> f16, dt_proj GEMM: [TOK,DTR] x [DI,DTR]^T -> dtraw [TOK,DI]
  cast_slice_kernel<<<TOK, 128, 0, stream>>>(zxbcdt, PROJ, ODT, DTR, dt16);
  gemm(dt16, w_dt16, dtraw, nullptr, TOK, DI, DTR);

  // 5) repeat_kv + causal depthwise conv + silu
  conv_silu_kernel<<<dim3(TOK, DI / 256), 256, 0, stream>>>(zxbcdt, conv_w, conv_b, xconv);

  // 6) selective scan (+D skip, silu(z) gate) -> y16 [TOK,DI]
  scan_kernel<<<BATCH * 16, 256, 0, stream>>>(zxbcdt, dtraw, xconv, A_log,
                                              dt_proj_b, D_param, y16);

  // 7) out_proj + residual(hidden) -> residb [TOK,DM]
  gemm(y16, w_out16, residb, hidden, TOK, DM, DI);

  // 8) rmsnorm(residb, ln2) -> f16
  rmsnorm_cast_kernel<<<TOK, 256, 0, stream>>>(residb, ln2, h2_16, DM);

  // 9) MLP: fused gate|up GEMM (N = 2*INTER), silu*mul, down GEMM + residual -> d_out
  gemm(h2_16, w_gu16, gu, nullptr, TOK, 2 * INTER, DM);
  silu_mul_cast_kernel<<<dim3((unsigned)(((size_t)TOK * INTER) / 256)), 256, 0, stream>>>(gu, act16);
  gemm(act16, w_down16, out, residb, TOK, DM, INTER);
}
